// MPDense_22926535426457
// MI455X (gfx1250) — compile-verified, matrix operand order checked by reading, not run on hardware
//
#include <hip/hip_runtime.h>

typedef __attribute__((ext_vector_type(16))) _Float16 v16h;
typedef __attribute__((ext_vector_type(8)))  float    v8f;
typedef __attribute__((ext_vector_type(4)))  float    v4f;
typedef __attribute__((ext_vector_type(2)))  float    v2f;

#define B_DIM 16384
#define D_DIM 2048
#define U_DIM 256

#define ROWS_PER_BLOCK 128      // 8 waves * 16 rows each
#define KT 512                  // K staged per LDS refill
#define NCHUNK (KT / 32)        // 16 chunks of K=32

// out[b,u] = sum_d x16[b,d]*w16[d,u]*(mp==0) + sum_d x[b,d]*w[d,u]*(mp==1) + bias[u]
__global__ __launch_bounds__(256)
void mpdense_wmma_kernel(const float* __restrict__ x,
                         const float* __restrict__ w,
                         const float* __restrict__ bias,
                         const int*   __restrict__ mp,
                         float* __restrict__ out)
{
    // Fragment-major staged B operands (masked weights), rebuilt per 512-K slab.
    __shared__ __align__(32) _Float16 sB16[NCHUNK * 512];  // 16 KB
    __shared__ __align__(16) float    sB32[NCHUNK * 512];  // 32 KB

    const int tid  = threadIdx.x;
    const int lane = tid & 31;
    const int wave = tid >> 5;

    const int tileN  = blockIdx.x & 15;     // 256/16 = 16 column tiles
    const int rowBlk = blockIdx.x >> 4;
    const int gn0    = tileN * 16;

    const int n    = lane & 15;
    const int hiL  = lane >> 4;             // 0: lanes 0-15, 1: lanes 16-31
    const int kb16 = hiL * 8;               // fp16 A-fragment K base per ISA layout
    const int kb32 = hiL * 2;               // fp32 A-fragment K base per ISA layout

    // A-matrix row for this lane (A layout: lanes 0-15 M=0..15, lanes 16-31 M=0..15)
    const int m = rowBlk * ROWS_PER_BLOCK + wave * 16 + n;

    v8f acc = {};

    for (int kt = 0; kt < D_DIM; kt += KT) {
        __syncthreads();
        // ---- cooperative fill: build masked-B WMMA fragments in LDS ----
        // 512 K-values x 16 cols = 8192 elements; 256 threads x 32 iters, no divergence.
        for (int t = tid; t < KT * 16; t += 256) {
            const int kl = t >> 4;          // 0..511 (K within slab)
            const int nn = t & 15;          // column within tile
            const int k  = kt + kl;
            const float wv = w[k * U_DIM + gn0 + nn];
            const int   mv = mp[k * U_DIM + gn0 + nn];
            const _Float16 w16 = (mv == 0) ? (_Float16)wv : (_Float16)0.0f;
            const float    w32 = (mv == 1) ? wv : 0.0f;

            const int c  = kl >> 5;         // chunk index within slab
            const int kk = kl & 31;         // K within chunk

            // fp16 B fragment inverse map (16-bit 32x16 B, wave32):
            // lane L holds K = kbase + (h&7) + (h>=8 ? 16:0), kbase = (L>=16)?8:0
            {
                const int r = kk & 15;
                const int L = nn + ((r >= 8) ? 16 : 0);
                const int h = (r & 7) + ((kk >= 16) ? 8 : 0);
                sB16[c * 512 + L * 16 + h] = w16;
            }
            // fp32 B fragment inverse map (32-bit 4x16 B per sub-chunk j):
            // lane L, sub-chunk j holds K = 4j + ((L>=16)?2:0) + {0,1}
            {
                const int j = kk >> 2;
                const int q = kk & 3;
                const int L = nn + ((q >= 2) ? 16 : 0);
                sB32[c * 512 + L * 16 + j * 2 + (q & 1)] = w32;
            }
        }
        __syncthreads();

        // ---- per-wave MMA over staged chunks, A streamed from global ----
        const float* xr = x + (size_t)m * D_DIM + kt;
        for (int c = 0; c < NCHUNK; ++c) {
            const int k0 = c * 32;

            // fp16 A fragment: lane reads 8 consecutive floats at kb16 and at 16+kb16
            v4f lo0 = *(const v4f*)(xr + k0 + kb16);
            v4f lo1 = *(const v4f*)(xr + k0 + kb16 + 4);
            v4f hi0 = *(const v4f*)(xr + k0 + 16 + kb16);
            v4f hi1 = *(const v4f*)(xr + k0 + 16 + kb16 + 4);
            v16h a16;
#pragma unroll
            for (int i = 0; i < 4; ++i) {
                a16[i]      = (_Float16)lo0[i];
                a16[4 + i]  = (_Float16)lo1[i];
                a16[8 + i]  = (_Float16)hi0[i];
                a16[12 + i] = (_Float16)hi1[i];
            }
            v16h b16 = *(const v16h*)(&sB16[c * 512 + lane * 16]);
            acc = __builtin_amdgcn_wmma_f32_16x16x32_f16(
                false, a16, false, b16, (short)0, acc, false, false);

            // fp32 bucket: 8 x v_wmma_f32_16x16x4_f32 over the same 32 K values
#pragma unroll
            for (int j = 0; j < 8; ++j) {
                v2f a32 = *(const v2f*)(xr + k0 + j * 4 + kb32);
                v2f b32 = *(const v2f*)(&sB32[c * 512 + lane * 16 + j * 2]);
                acc = __builtin_amdgcn_wmma_f32_16x16x4_f32(
                    false, a32, false, b32, (short)0, acc, false, false);
            }
        }
    }

    // ---- epilogue: bias + store (C/D layout: VGPR r -> M = r + 8*hiL, N = lane&15) ----
    const float bb = bias[gn0 + n];
    const int rowBase = rowBlk * ROWS_PER_BLOCK + wave * 16 + hiL * 8;
#pragma unroll
    for (int r = 0; r < 8; ++r) {
        out[(size_t)(rowBase + r) * U_DIM + gn0 + n] = acc[r] + bb;
    }
}

extern "C" void kernel_launch(void* const* d_in, const int* in_sizes, int n_in,
                              void* d_out, int out_size, void* d_ws, size_t ws_size,
                              hipStream_t stream) {
    (void)in_sizes; (void)n_in; (void)d_ws; (void)ws_size; (void)out_size;
    const float* x    = (const float*)d_in[0];
    const float* w    = (const float*)d_in[1];
    const float* bias = (const float*)d_in[2];
    const int*   mp   = (const int*)d_in[3];
    float* out = (float*)d_out;

    dim3 grid((B_DIM / ROWS_PER_BLOCK) * (U_DIM / 16));  // 128 * 16 = 2048 blocks
    dim3 block(256);                                     // 8 waves (wave32)
    mpdense_wmma_kernel<<<grid, block, 0, stream>>>(x, w, bias, mp, out);
}
